// MyEfficientMultiHeadAttention_35150012350466
// MI455X (gfx1250) — matrix operand order repaired and checked by reading, not compiled, not run on hardware
//
#include <hip/hip_runtime.h>
#include <hip/hip_bf16.h>

// MHA forward for MI455X (gfx1250, wave32, WMMA f32_16x16x32_f16).
// B=2 S=2048 E=1024 H=16 D=64.
// Pipeline:
//   1) proj_gemm   x3 : Q,K (f16, [B,H,S,D]) and V^T (f16, [B,H,D,S]) in ws
//   2) attn_scores    : per (b,h, 16-row strip): S = QK^T/8 via WMMA,
//                       strip softmax in 128KB dynamic LDS, write fp32 P to d_out
//   3) attn_v         : attn = P @ V via WMMA (P fp32 -> f16 on load, V^T tile
//                       staged with global_load_async_to_lds_b128)
//   4) out_proj       : output = attn @ Wo^T + bo (attn tile staged async)
// ~69 GFLOP of GEMM through v_wmma; dominant traffic is the mandatory 536MB
// attn_weights write + one re-read for P@V (~46us at 23.3 TB/s), so the P
// stream gets global_prefetch_b8 hints.

#define BB 2
#define SS 2048
#define EE 1024
#define HH 16
#define DD 64
#define MM (BB * SS) // 4096

typedef __attribute__((ext_vector_type(16))) _Float16 v16h;
typedef __attribute__((ext_vector_type(8))) float v8f;

union FragH { v16h v; uint4 q[2]; };

__device__ inline v8f wmma_f16(v16h a, v16h b, v8f c) {
  return __builtin_amdgcn_wmma_f32_16x16x32_f16(false, a, false, b, (short)0, c,
                                                false, false);
}

// Async 16B global -> LDS copy (CDNA5 GLOBAL_LOAD_ASYNC_TO_LDS_B128, ASYNCcnt).
// LDS byte offset = low 32 bits of the generic pointer (aperture is in the
// high word on gfx1250).
__device__ inline void async_copy16(const _Float16* gsrc, _Float16* ldst) {
  unsigned l = (unsigned)(size_t)ldst;
  asm volatile("global_load_async_to_lds_b128 %0, %1, off"
               :
               : "v"(l), "v"(gsrc)
               : "memory");
}
__device__ inline void wait_async0() {
  asm volatile("s_wait_asynccnt 0x0" ::: "memory");
}

// A fragment (16xK strip, row-major, ld in halves). ISA 16-bit A layout:
// lanes 0-15: VGPR0-3 = K[kk+0..7], VGPR4-7 = K[kk+16..23]; lanes 16-31: +8.
__device__ inline v16h load_fragA(const _Float16* base, int ld, int lane, int kk) {
  const int m = lane & 15, g = lane >> 4;
  const _Float16* p = base + m * ld + kk + g * 8;
  FragH f;
  f.q[0] = *(const uint4*)p;
  f.q[1] = *(const uint4*)(p + 16);
  return f.v;
}

// B fragment from [N][K] row-major strip. 16-bit B layout: lane n=lane&15,
// lanes 0-15: K[kk+0..7], K[kk+16..23]; lanes 16-31: +8 (matches A).
__device__ inline v16h load_fragB(const _Float16* base, int ld, int lane, int kk) {
  const int n = lane & 15, g = lane >> 4;
  const _Float16* p = base + n * ld + kk + g * 8;
  FragH f;
  f.q[0] = *(const uint4*)p;
  f.q[1] = *(const uint4*)(p + 16);
  return f.v;
}

// ---------------- Kernel 1: QKV projections ----------------
// out[m,n] = sum_k X[m,k]*W[n,k] + bias[n], stored f16 split-head
// (transpose_v=0 -> [B,H,S,D], transpose_v=1 -> [B,H,D,S]).
__global__ __launch_bounds__(256) void proj_gemm_kernel(
    const float* __restrict__ X, const float* __restrict__ W,
    const float* __restrict__ bias, _Float16* __restrict__ out,
    int transpose_v) {
  __shared__ _Float16 As[128 * 40];
  __shared__ _Float16 Bs[64 * 40];
  const int tid = threadIdx.x;
  const int lane = tid & 31, wave = tid >> 5;
  const int wm = wave & 3, wn = wave >> 2;
  const int m0 = blockIdx.x * 128, n0 = blockIdx.y * 64;

  const v8f vzero = {0.f, 0.f, 0.f, 0.f, 0.f, 0.f, 0.f, 0.f};
  v8f acc[2][2] = {{vzero, vzero}, {vzero, vzero}};

  for (int k0 = 0; k0 < EE; k0 += 32) {
    // A tile 128x32 fp32 -> f16 LDS (4 quads / thread)
#pragma unroll
    for (int i = 0; i < 4; ++i) {
      int q = tid * 4 + i;
      int r = q >> 3, c = (q & 7) * 4;
      const float* src = X + (size_t)(m0 + r) * EE + k0 + c;
      float4 xv = *(const float4*)src;
      if (i == 0 && k0 + 32 < EE) __builtin_prefetch(src + 32, 0, 3);
      _Float16* d = &As[r * 40 + c];
      d[0] = (_Float16)xv.x; d[1] = (_Float16)xv.y;
      d[2] = (_Float16)xv.z; d[3] = (_Float16)xv.w;
    }
    // B tile 64x32 fp32 -> f16 LDS (2 quads / thread)
#pragma unroll
    for (int i = 0; i < 2; ++i) {
      int q = tid * 2 + i;
      int r = q >> 3, c = (q & 7) * 4;
      const float* src = W + (size_t)(n0 + r) * EE + k0 + c;
      float4 wv = *(const float4*)src;
      if (i == 0 && k0 + 32 < EE) __builtin_prefetch(src + 32, 0, 3);
      _Float16* d = &Bs[r * 40 + c];
      d[0] = (_Float16)wv.x; d[1] = (_Float16)wv.y;
      d[2] = (_Float16)wv.z; d[3] = (_Float16)wv.w;
    }
    __syncthreads();
#pragma unroll
    for (int i = 0; i < 2; ++i) {
      v16h a = load_fragA(&As[(wm * 32 + i * 16) * 40], 40, lane, 0);
#pragma unroll
      for (int j = 0; j < 2; ++j) {
        v16h b = load_fragB(&Bs[(wn * 32 + j * 16) * 40], 40, lane, 0);
        acc[i][j] = wmma_f16(a, b, acc[i][j]);
      }
    }
    __syncthreads();
  }

  const int g = lane >> 4, ln = lane & 15;
#pragma unroll
  for (int i = 0; i < 2; ++i) {
#pragma unroll
    for (int j = 0; j < 2; ++j) {
      int n = n0 + wn * 32 + j * 16 + ln;
      float bn = bias[n];
      int h = n >> 6, d = n & 63;
#pragma unroll
      for (int r = 0; r < 8; ++r) {
        int m = m0 + wm * 32 + i * 16 + g * 8 + r;
        int b = m >> 11, s = m & (SS - 1);
        float v = acc[i][j][r] + bn;
        size_t addr = transpose_v
                          ? ((size_t)(b * HH + h) * DD + d) * SS + s
                          : ((size_t)(b * HH + h) * SS + s) * DD + d;
        out[addr] = (_Float16)v;
      }
    }
  }
}

// ---------------- Kernel 2: scores + softmax ----------------
// One workgroup per (b*h, 16-row strip). Scores strip (16 x 2048 fp32) in
// dynamic LDS; Q/K fragments loaded directly from global f16.
__global__ __launch_bounds__(256) void attn_scores_kernel(
    const _Float16* __restrict__ Q, const _Float16* __restrict__ K,
    float* __restrict__ P) {
  extern __shared__ float sc[]; // 16 * 2048 floats = 128 KB
  const int bh = blockIdx.y;
  const int row0 = blockIdx.x * 16;
  const int tid = threadIdx.x, lane = tid & 31, wave = tid >> 5;
  const _Float16* qb = Q + ((size_t)bh * SS + row0) * DD;
  const _Float16* kb = K + (size_t)bh * SS * DD;

  v16h aF0 = load_fragA(qb, DD, lane, 0);
  v16h aF1 = load_fragA(qb, DD, lane, 32);
  const int g = lane >> 4, ln = lane & 15;

  for (int ct = 0; ct < 16; ++ct) {
    int n0 = wave * 256 + ct * 16;
    if (ct < 15) // prefetch next K tile rows for this lane
      __builtin_prefetch(kb + (size_t)(n0 + 16 + ln) * DD, 0, 3);
    v8f acc = {0.f, 0.f, 0.f, 0.f, 0.f, 0.f, 0.f, 0.f};
    v16h b0 = load_fragB(kb + (size_t)n0 * DD, DD, lane, 0);
    acc = wmma_f16(aF0, b0, acc);
    v16h b1 = load_fragB(kb + (size_t)n0 * DD, DD, lane, 32);
    acc = wmma_f16(aF1, b1, acc);
#pragma unroll
    for (int r = 0; r < 8; ++r) {
      int mrow = g * 8 + r;
      sc[mrow * SS + n0 + ln] = acc[r] * 0.125f; // 1/sqrt(64)
    }
  }
  __syncthreads();

  // Softmax: 16 lanes per row (half-wave groups).
  const int row = tid >> 4, li = tid & 15;
  float* srow = sc + row * SS;
  float mx = -3.4e38f;
  for (int j = li; j < SS; j += 16) mx = fmaxf(mx, srow[j]);
#pragma unroll
  for (int off = 8; off; off >>= 1) mx = fmaxf(mx, __shfl_xor(mx, off, 16));
  float sum = 0.f;
  for (int j = li; j < SS; j += 16) {
    float e = __expf(srow[j] - mx);
    srow[j] = e;
    sum += e;
  }
#pragma unroll
  for (int off = 8; off; off >>= 1) sum += __shfl_xor(sum, off, 16);
  float inv = 1.0f / sum;
  float* prow = P + ((size_t)bh * SS + row0 + row) * SS;
  for (int j = li; j < SS; j += 16) prow[j] = srow[j] * inv;
}

// ---------------- Kernel 3: attn = P @ V ----------------
// Per (b,h): M=2048 (s), N=64 (d), K=2048 (t). A = P fp32 (converted),
// B = V^T f16 staged with async global->LDS copies.
__global__ __launch_bounds__(256) void attn_v_kernel(
    const float* __restrict__ P, const _Float16* __restrict__ VT,
    _Float16* __restrict__ AM) {
  __shared__ _Float16 As[128 * 40];
  __shared__ _Float16 Bs[64 * 40];
  const int bh = blockIdx.y;
  const int b = bh >> 4, h = bh & 15;
  const int m0 = blockIdx.x * 128;
  const int tid = threadIdx.x, lane = tid & 31, wave = tid >> 5;
  const int wm = wave & 3, wn = wave >> 2;
  const float* pb = P + ((size_t)bh * SS + m0) * SS;
  const _Float16* vb = VT + (size_t)bh * DD * SS;

  const v8f vzero = {0.f, 0.f, 0.f, 0.f, 0.f, 0.f, 0.f, 0.f};
  v8f acc[2][2] = {{vzero, vzero}, {vzero, vzero}};

  const int bd = tid >> 2, bc = (tid & 3) * 8; // async B-tile copy coords

  for (int k0 = 0; k0 < SS; k0 += 32) {
    // B tile 64x32 f16 from V^T: pure copy -> async global->LDS (no VGPR trip)
    async_copy16(vb + (size_t)bd * SS + k0 + bc, &Bs[bd * 40 + bc]);

    // A tile 128x32 from P fp32 (the 536MB stream: prefetch next tile)
#pragma unroll
    for (int i = 0; i < 4; ++i) {
      int q = tid * 4 + i;
      int r = q >> 3, c = (q & 7) * 4;
      const float* src = pb + (size_t)r * SS + k0 + c;
      float4 pv = *(const float4*)src;
      if (i == 0 && k0 + 32 < SS) __builtin_prefetch(src + 32, 0, 3);
      _Float16* d = &As[r * 40 + c];
      d[0] = (_Float16)pv.x; d[1] = (_Float16)pv.y;
      d[2] = (_Float16)pv.z; d[3] = (_Float16)pv.w;
    }
    wait_async0();
    __syncthreads();
#pragma unroll
    for (int i = 0; i < 2; ++i) {
      v16h a = load_fragA(&As[(wm * 32 + i * 16) * 40], 40, lane, 0);
#pragma unroll
      for (int j = 0; j < 2; ++j) {
        v16h bfr = load_fragB(&Bs[(wn * 32 + j * 16) * 40], 40, lane, 0);
        acc[i][j] = wmma_f16(a, bfr, acc[i][j]);
      }
    }
    __syncthreads();
  }

  const int g = lane >> 4, ln = lane & 15;
#pragma unroll
  for (int i = 0; i < 2; ++i) {
#pragma unroll
    for (int j = 0; j < 2; ++j) {
      int d = wn * 32 + j * 16 + ln;
#pragma unroll
      for (int r = 0; r < 8; ++r) {
        int s = m0 + wm * 32 + i * 16 + g * 8 + r;
        size_t addr = ((size_t)(b * SS + s)) * EE + h * 64 + d;
        AM[addr] = (_Float16)acc[i][j][r];
      }
    }
  }
}

// ---------------- Kernel 4: output projection ----------------
// out[m,n] = sum_k AM[m,k]*Wo[n,k] + bo[n], fp32 out. A tile staged async.
__global__ __launch_bounds__(256) void out_proj_kernel(
    const _Float16* __restrict__ AM, const float* __restrict__ Wo,
    const float* __restrict__ bo, float* __restrict__ out) {
  __shared__ _Float16 As[128 * 40];
  __shared__ _Float16 Bs[64 * 40];
  const int tid = threadIdx.x, lane = tid & 31, wave = tid >> 5;
  const int wm = wave & 3, wn = wave >> 2;
  const int m0 = blockIdx.x * 128, n0 = blockIdx.y * 64;

  const v8f vzero = {0.f, 0.f, 0.f, 0.f, 0.f, 0.f, 0.f, 0.f};
  v8f acc[2][2] = {{vzero, vzero}, {vzero, vzero}};

  for (int k0 = 0; k0 < EE; k0 += 32) {
    // A tile 128x32 f16: pure copy -> async global->LDS (2x 16B per thread)
#pragma unroll
    for (int i = 0; i < 2; ++i) {
      int c = tid * 2 + i;
      int r = c >> 2, o = (c & 3) * 8;
      async_copy16(AM + (size_t)(m0 + r) * EE + k0 + o, &As[r * 40 + o]);
    }
    // B tile 64x32 from Wo fp32
#pragma unroll
    for (int i = 0; i < 2; ++i) {
      int q = tid * 2 + i;
      int r = q >> 3, c = (q & 7) * 4;
      const float* src = Wo + (size_t)(n0 + r) * EE + k0 + c;
      float4 wv = *(const float4*)src;
      if (i == 0 && k0 + 32 < EE) __builtin_prefetch(src + 32, 0, 3);
      _Float16* d = &Bs[r * 40 + c];
      d[0] = (_Float16)wv.x; d[1] = (_Float16)wv.y;
      d[2] = (_Float16)wv.z; d[3] = (_Float16)wv.w;
    }
    wait_async0();
    __syncthreads();
#pragma unroll
    for (int i = 0; i < 2; ++i) {
      v16h a = load_fragA(&As[(wm * 32 + i * 16) * 40], 40, lane, 0);
#pragma unroll
      for (int j = 0; j < 2; ++j) {
        v16h b = load_fragB(&Bs[(wn * 32 + j * 16) * 40], 40, lane, 0);
        acc[i][j] = wmma_f16(a, b, acc[i][j]);
      }
    }
    __syncthreads();
  }

  const int g = lane >> 4, ln = lane & 15;
#pragma unroll
  for (int i = 0; i < 2; ++i) {
#pragma unroll
    for (int j = 0; j < 2; ++j) {
      int n = n0 + wn * 32 + j * 16 + ln;
      float bn = bo[n];
#pragma unroll
      for (int r = 0; r < 8; ++r) {
        int m = m0 + wm * 32 + i * 16 + g * 8 + r;
        out[(size_t)m * EE + n] = acc[i][j][r] + bn;
      }
    }
  }
}

extern "C" void kernel_launch(void* const* d_in, const int* in_sizes, int n_in,
                              void* d_out, int out_size, void* d_ws,
                              size_t ws_size, hipStream_t stream) {
  const float* x = (const float*)d_in[0];
  const float* Wq = (const float*)d_in[1];
  const float* bq = (const float*)d_in[2];
  const float* Wk = (const float*)d_in[3];
  const float* bk = (const float*)d_in[4];
  const float* Wv = (const float*)d_in[5];
  const float* bv = (const float*)d_in[6];
  const float* Wo = (const float*)d_in[7];
  const float* bo = (const float*)d_in[8];

  float* out = (float*)d_out;               // [B,S,E]
  float* attw = out + (size_t)BB * SS * EE; // [B,H,S,S]

  const size_t QSZ = (size_t)MM * EE; // halves per tensor
  _Float16* ws = (_Float16*)d_ws;
  _Float16* Qw = ws;           // [B,H,S,D] f16
  _Float16* Kw = ws + QSZ;     // [B,H,S,D] f16
  _Float16* VT = ws + 2 * QSZ; // [B,H,D,S] f16
  _Float16* AM = ws + 3 * QSZ; // [B,S,E]   f16

  dim3 blk(256);
  dim3 g1(MM / 128, EE / 64);
  proj_gemm_kernel<<<g1, blk, 0, stream>>>(x, Wq, bq, Qw, 0);
  proj_gemm_kernel<<<g1, blk, 0, stream>>>(x, Wk, bk, Kw, 0);
  proj_gemm_kernel<<<g1, blk, 0, stream>>>(x, Wv, bv, VT, 1);

  dim3 g2(SS / 16, BB * HH);
  attn_scores_kernel<<<g2, blk, 16 * SS * sizeof(float), stream>>>(Qw, Kw, attw);

  dim3 g3(SS / 128, BB * HH);
  attn_v_kernel<<<g3, blk, 0, stream>>>(attw, VT, AM);

  dim3 g4(MM / 128, EE / 64);
  out_proj_kernel<<<g4, blk, 0, stream>>>(AM, Wo, bo, out);
}
